// SelectiveSearch_65798898975436
// MI455X (gfx1250) — compile-verified, matrix operand order checked by reading, not run on hardware
//
#include <hip/hip_runtime.h>
#include <hip/hip_bf16.h>
#include <math.h>

// ---------------- problem constants (match setup_inputs) ----------------
#define CIMG   3
#define HH     1024
#define WW     1024
#define SPIX   (HH * WW)          // 1048576
#define NSEG   5000
#define CB     25                 // color bins
#define TB     10                 // texture bins
#define CH_NB  (CIMG * CB)        // 75 bins/segment (color)
#define TH_NB  (CIMG * 8 * TB)    // 240 bins/segment (texture)

// d_out layout (floats, concat of tuple in return order)
#define OUT_XYWH   0              // 5000*4
#define OUT_SIZES  20000          // 5000
#define OUT_CHIST  25000          // 375000
#define OUT_THIST  400000         // 1200000
#define OUT_TOTAL  1600000

typedef __attribute__((ext_vector_type(16))) _Float16 v16h;
typedef __attribute__((ext_vector_type(8)))  float    v8f;

// ---------------- order-preserving float<->uint encode for atomic min/max ----
__device__ __forceinline__ unsigned encf(float f) {
  int i = __float_as_int(f);
  return (i < 0) ? ~(unsigned)i : ((unsigned)i | 0x80000000u);
}
__device__ __forceinline__ float decf(unsigned u) {
  int i = (u & 0x80000000u) ? (int)(u & 0x7fffffffu) : ~(int)u;
  return __int_as_float(i);
}

// ---------------- init: zero d_out, init bbox/count/minmax scratch ----------
__global__ void k_init(float* out, int* iw, unsigned* mm) {
  for (int i = blockIdx.x * blockDim.x + threadIdx.x; i < OUT_TOTAL;
       i += blockDim.x * gridDim.x) {
    out[i] = 0.0f;
    if (i < NSEG) {
      iw[i]            = 0x7fffffff;      // xmin
      iw[NSEG + i]     = 0x7fffffff;      // ymin
      iw[2 * NSEG + i] = (int)0x80000000; // xmax
      iw[3 * NSEG + i] = (int)0x80000000; // ymax
      iw[4 * NSEG + i] = 0;               // count
    }
    if (i < 48) mm[i] = (i < 24) ? 0xFFFFFFFFu : 0u;
  }
}

// ---------------- per-segment bbox + size ----------------------------------
__global__ void k_bbox(const int* __restrict__ reg, int* iw) {
  for (int i = blockIdx.x * blockDim.x + threadIdx.x; i < SPIX;
       i += blockDim.x * gridDim.x) {
    int seg = reg[i];
    int x = i % WW, y = i / WW;
    atomicMin(&iw[seg], x);
    atomicMin(&iw[NSEG + seg], y);
    atomicMax(&iw[2 * NSEG + seg], x);
    atomicMax(&iw[3 * NSEG + seg], y);
    atomicAdd(&iw[4 * NSEG + seg], 1);
  }
}

// ---------------- Scharr (cross-correlation, zero pad), per channel --------
// kx = [[-3,0,3],[10,0,10],[-3,0,3]]; gy uses kx^T. out layout [c][2][H][W].
__global__ void k_scharr(const float* __restrict__ in, float* __restrict__ out,
                         int C_, int H, int W) {
  int HWp = H * W;
  int total = C_ * HWp;
  int stride = blockDim.x * gridDim.x;
  for (int i = blockIdx.x * blockDim.x + threadIdx.x; i < total; i += stride) {
    __builtin_prefetch(&in[min(i + stride, total - 1)], 0, 0);
    int c = i / HWp, p = i - c * HWp;
    int y = p / W, x = p - y * W;
    const float* base = in + c * HWp;
    float n[3][3];
#pragma unroll
    for (int dy = -1; dy <= 1; ++dy)
#pragma unroll
      for (int dx = -1; dx <= 1; ++dx) {
        int yy = y + dy, xx = x + dx;
        bool ok = (yy >= 0) & (yy < H) & (xx >= 0) & (xx < W);
        n[dy + 1][dx + 1] = ok ? base[yy * W + xx] : 0.0f;
      }
    float gx = -3.0f * n[0][0] + 3.0f * n[0][2] + 10.0f * n[1][0] +
               10.0f * n[1][2] - 3.0f * n[2][0] + 3.0f * n[2][2];
    float gy = -3.0f * n[0][0] + 10.0f * n[0][1] - 3.0f * n[0][2] +
               3.0f * n[2][0] + 10.0f * n[2][1] + 3.0f * n[2][2];
    out[(c * 2 + 0) * HWp + p] = gx;
    out[(c * 2 + 1) * HWp + p] = gy;
  }
}

// ---------------- nearest-neighbor rotate (full output grid) ---------------
__global__ void k_rotate(const float* __restrict__ in, float* __restrict__ out,
                         int nmaps, int inH, int inW, int outH, int outW,
                         float cF, float sF) {
  int HWi = inH * inW, HWo = outH * outW;
  float ncx = (outW - 1) * 0.5f, ncy = (outH - 1) * 0.5f;
  float cx = (inW - 1) * 0.5f, cy = (inH - 1) * 0.5f;
  int total = nmaps * HWo;
  for (int i = blockIdx.x * blockDim.x + threadIdx.x; i < total;
       i += blockDim.x * gridDim.x) {
    int m = i / HWo, p = i - m * HWo;
    int y = p / outW, x = p - y * outW;
    float fx = (float)x - ncx, fy = (float)y - ncy;
    float sx = cF * fx - sF * fy + cx;
    float sy = sF * fx + cF * fy + cy;
    int ix = min(max((int)rintf(sx), 0), inW - 1);
    int iy = min(max((int)rintf(sy), 0), inH - 1);
    bool valid = (sx >= -0.5f) & (sx <= (float)inW - 0.5f) &
                 (sy >= -0.5f) & (sy <= (float)inH - 0.5f);
    out[m * HWo + p] = valid ? in[m * HWi + iy * inW + ix] : 0.0f;
  }
}

// -------- inverse rotate evaluated only on the cropped 1024x1024 window ----
__global__ void k_rotback_crop(const float* __restrict__ in,
                               float* __restrict__ out, int nmaps, int inH,
                               int inW, int bigH, int bigW, int sx0, int sy0,
                               float cF, float sF) {
  int HWi = inH * inW;
  float ncx = (bigW - 1) * 0.5f, ncy = (bigH - 1) * 0.5f;
  float cx = (inW - 1) * 0.5f, cy = (inH - 1) * 0.5f;
  int total = nmaps * SPIX;
  for (int i = blockIdx.x * blockDim.x + threadIdx.x; i < total;
       i += blockDim.x * gridDim.x) {
    int m = i / SPIX, p = i - m * SPIX;
    int y = p / WW, x = p - y * WW;
    float fx = (float)(x + sx0) - ncx, fy = (float)(y + sy0) - ncy;
    float sx = cF * fx - sF * fy + cx;
    float sy = sF * fx + cF * fy + cy;
    int ix = min(max((int)rintf(sx), 0), inW - 1);
    int iy = min(max((int)rintf(sy), 0), inH - 1);
    bool valid = (sx >= -0.5f) & (sx <= (float)inW - 0.5f) &
                 (sy >= -0.5f) & (sy <= (float)inH - 0.5f);
    out[m * SPIX + p] = valid ? in[m * HWi + iy * inW + ix] : 0.0f;
  }
}

// map m = c*8 + k: k0/1 max(g), k2/3 min(g), k4/5 max(grot), k6/7 min(grot)
__device__ __forceinline__ float deriv_val(const float* G, const float* GRC,
                                           int m, int p) {
  int c = m >> 3, k = m & 7;
  const float* src = ((k < 4) ? G : GRC) + (c * 2 + (k & 1)) * SPIX;
  float v = src[p];
  return (k & 2) ? fminf(v, 0.0f) : fmaxf(v, 0.0f);
}

// ---------------- per-map min/max via LDS reduce + global atomic -----------
__global__ void k_minmax(const float* __restrict__ G,
                         const float* __restrict__ GRC, unsigned* mn,
                         unsigned* mx) {
  __shared__ float smin[256], smax[256];
  int m = blockIdx.y;
  float lmin = 3.4e38f, lmax = -3.4e38f;
  for (int i = blockIdx.x * blockDim.x + threadIdx.x; i < SPIX;
       i += blockDim.x * gridDim.x) {
    float v = deriv_val(G, GRC, m, i);
    lmin = fminf(lmin, v);
    lmax = fmaxf(lmax, v);
  }
  int t = threadIdx.x;
  smin[t] = lmin; smax[t] = lmax;
  __syncthreads();
  for (int s = 128; s > 0; s >>= 1) {
    if (t < s) {
      smin[t] = fminf(smin[t], smin[t + s]);
      smax[t] = fmaxf(smax[t], smax[t + s]);
    }
    __syncthreads();
  }
  if (t == 0) {
    atomicMin(&mn[m], encf(smin[0]));
    atomicMax(&mx[m], encf(smax[0]));
  }
}

// ---------------- color histogram ------------------------------------------
__global__ void k_chist(const float* __restrict__ img,
                        const int* __restrict__ reg, float* ch) {
  for (int i = blockIdx.x * blockDim.x + threadIdx.x; i < SPIX;
       i += blockDim.x * gridDim.x) {
    int seg = reg[i];
#pragma unroll
    for (int c = 0; c < CIMG; ++c) {
      float v = img[c * SPIX + i];
      int q = min(max((int)(v * (float)(CB - 1)), 0), CB - 1);
      atomicAdd(&ch[seg * CH_NB + c * CB + q], 1.0f);
    }
  }
}

// ---------------- texture histogram ----------------------------------------
__global__ void k_thist(const float* __restrict__ G,
                        const float* __restrict__ GRC,
                        const int* __restrict__ reg,
                        const unsigned* __restrict__ mnmx, float* th) {
  __shared__ float smn[24], smx[24];
  if (threadIdx.x < 24) {
    smn[threadIdx.x] = decf(mnmx[threadIdx.x]);
    smx[threadIdx.x] = decf(mnmx[24 + threadIdx.x]);
  }
  __syncthreads();
  for (int i = blockIdx.x * blockDim.x + threadIdx.x; i < SPIX;
       i += blockDim.x * gridDim.x) {
    int seg = reg[i];
#pragma unroll
    for (int m = 0; m < 24; ++m) {
      float v = deriv_val(G, GRC, m, i);
      float nrm = (v - smn[m]) / (smx[m] - smn[m]); // matches jnp division
      int q = min(max((int)(nrm * (float)(TB - 1)), 0), TB - 1);
      int c = m >> 3, k = m & 7;
      atomicAdd(&th[seg * TH_NB + c * (8 * TB) + k * TB + q], 1.0f);
    }
  }
}

// ---------------- write xywh + sizes ---------------------------------------
__global__ void k_writeseg(const int* __restrict__ iw, float* out) {
  int s = blockIdx.x * blockDim.x + threadIdx.x;
  if (s >= NSEG) return;
  int xmn = iw[s], ymn = iw[NSEG + s], xmx = iw[2 * NSEG + s],
      ymx = iw[3 * NSEG + s];
  out[OUT_XYWH + s * 4 + 0] = (float)xmn;
  out[OUT_XYWH + s * 4 + 1] = (float)ymn;
  out[OUT_XYWH + s * 4 + 2] = (float)(xmx - xmn);
  out[OUT_XYWH + s * 4 + 3] = (float)(ymx - ymn);
  out[OUT_SIZES + s] = (float)iw[4 * NSEG + s];
}

// ---------------- histogram normalization via WMMA -------------------------
// Row-sums of a (16 seg x NB) tile: D = A(16x32 f16) * ones(32x16) + C, f32
// accumulate. Counts are integers < 2048 -> f16 exact -> sums exact.
// A striping per ISA 7.12.2 (16-bit A 16x32): for lane L (hi = L>=16), the
// lane's 16 halves are two contiguous 8-bin runs: [k0+8*hi, +8) and
// [k0+16+8*hi, +8), with row M = L&15.
//
// Loads use clamped always-in-range indices + post-load select, so they emit
// as a straight batch of global_load_b32 (one wait), no EXEC divergence, and
// EXEC stays all-ones through the WMMA as the ISA requires.
__global__ __launch_bounds__(32) void k_norm_wmma(float* hist, int NB) {
  __shared__ float sums[16];
  int lane = threadIdx.x;
  int segbase = blockIdx.x * 16;
  int seg = segbase + (lane & 15);
  bool segok = seg < NSEG;
  const float* row = hist + (size_t)(segok ? seg : (NSEG - 1)) * NB;
  bool hi = lane >= 16;

  v16h bmat;
#pragma unroll
  for (int e = 0; e < 16; ++e) bmat[e] = (_Float16)1.0f;

  v8f acc = {};
  int nch = (NB + 31) / 32;
  for (int kc = 0; kc < nch; ++kc) {
    int k0 = kc * 32 + (hi ? 8 : 0);
    float buf[16];
#pragma unroll
    for (int j = 0; j < 8; ++j) {       // unconditional, clamped loads
      buf[j]     = row[min(k0 + j, NB - 1)];
      buf[8 + j] = row[min(k0 + 16 + j, NB - 1)];
    }
    v16h a;
#pragma unroll
    for (int j = 0; j < 8; ++j) {       // mask via select, not branch
      bool okA = segok && (k0 + j) < NB;
      bool okB = segok && (k0 + 16 + j) < NB;
      a[j]     = (_Float16)(okA ? buf[j] : 0.0f);
      a[8 + j] = (_Float16)(okB ? buf[8 + j] : 0.0f);
    }
    acc = __builtin_amdgcn_wmma_f32_16x16x32_f16(
        /*neg_a=*/false, a, /*neg_b=*/false, bmat,
        /*c_mod=*/(short)0, acc, /*reuse_a=*/false, /*reuse_b=*/false);
  }
  // D layout: VGPR r, lanes 0-15 -> M=r; lanes 16-31 -> M=8+r (all N equal).
  if (lane == 0)
    for (int r = 0; r < 8; ++r) sums[r] = acc[r];
  if (lane == 16)
    for (int r = 0; r < 8; ++r) sums[8 + r] = acc[r];
  __syncthreads();
  for (int idx = lane; idx < 16 * NB; idx += 32) {
    int m = idx / NB;
    int s = segbase + m;
    if (s < NSEG) hist[s * NB + (idx - m * NB)] /= sums[m]; // 0/0 -> NaN == jax
  }
}

// ===========================================================================
extern "C" void kernel_launch(void* const* d_in, const int* in_sizes, int n_in,
                              void* d_out, int out_size, void* d_ws,
                              size_t ws_size, hipStream_t stream) {
  const float* img = (const float*)d_in[0];
  const int* reg = (const int*)d_in[1];
  float* out = (float*)d_out;

  // ---- rotation geometry, computed exactly like the Python reference ----
  const double thd = 0.7853981633974483096; // radians(45)
  const double cd = cos(thd), sd = sin(thd);
  const int nW1 = (int)ceil(fabs(WW * cd) + fabs(HH * sd)); // 1449
  const int nH1 = (int)ceil(fabs(WW * sd) + fabs(HH * cd)); // 1449
  const int nW2 = (int)ceil(fabs(nW1 * cd) + fabs(nH1 * sd)); // 2050
  const int nH2 = (int)ceil(fabs(nW1 * sd) + fabs(nH1 * cd)); // 2050
  // rotated_xywh(HH, WW, 45): bbox of the 4 rotated corners
  const double a = cd, b = sd;
  const double ccx = WW / 2.0, ccy = HH / 2.0;
  const double tx = (1 - a) * ccx - b * ccy, ty = b * ccx + (1 - a) * ccy;
  double px[4] = {0.0, (double)(WW - 1), 0.0, (double)(WW - 1)};
  double py[4] = {0.0, 0.0, (double)(HH - 1), (double)(HH - 1)};
  double mnx = 1e30, mxx = -1e30, mny = 1e30, mxy = -1e30;
  for (int i = 0; i < 4; ++i) {
    double xr = a * px[i] + b * py[i] + tx;
    double yr = -b * px[i] + a * py[i] + ty;
    mnx = fmin(mnx, xr); mxx = fmax(mxx, xr);
    mny = fmin(mny, yr); mxy = fmax(mxy, yr);
  }
  const int startx = (int)fmax(0.0, ((mxx - mnx) - WW) / 2.0); // 211
  const int starty = (int)fmax(0.0, ((mxy - mny) - HH) / 2.0); // 211

  // ---- workspace carve-up (floats) ----
  float* WS = (float*)d_ws;
  size_t off = 0;
  float* G = WS + off;                 // 6 * SPIX   (g: 3ch x 2dir)
  off += 6ull * SPIX;
  size_t szRot = 3ull * nH1 * nW1;     // rotated image
  float* IMGROT = WS + off;
  float* GRC = IMGROT;                 // reuse after IMGROT is dead (6*SPIX <= szRot)
  off += (szRot > 6ull * SPIX ? szRot : 6ull * SPIX);
  float* GROT = WS + off;              // 6 * nH1*nW1 (gradients of rotated)
  off += 6ull * (size_t)nH1 * nW1;
  int* IW = (int*)(WS + off);          // bbox mins/maxs/counts: 5*NSEG ints
  unsigned* MM = (unsigned*)(IW + 5 * NSEG); // 24 enc-min + 24 enc-max

  const int B = 256;
  k_init<<<2048, B, 0, stream>>>(out, IW, MM);
  k_bbox<<<4096, B, 0, stream>>>(reg, IW);
  k_scharr<<<4096, B, 0, stream>>>(img, G, CIMG, HH, WW);
  k_rotate<<<4096, B, 0, stream>>>(img, IMGROT, CIMG, HH, WW, nH1, nW1,
                                   (float)cd, (float)sd);
  k_scharr<<<4096, B, 0, stream>>>(IMGROT, GROT, CIMG, nH1, nW1);
  k_rotback_crop<<<4096, B, 0, stream>>>(GROT, GRC, 6, nH1, nW1, nH2, nW2,
                                         startx, starty, (float)cd,
                                         (float)(-sd));
  k_minmax<<<dim3(64, 24), B, 0, stream>>>(G, GRC, MM, MM + 24);
  k_chist<<<4096, B, 0, stream>>>(img, reg, out + OUT_CHIST);
  k_thist<<<4096, B, 0, stream>>>(G, GRC, reg, MM, out + OUT_THIST);
  k_writeseg<<<(NSEG + B - 1) / B, B, 0, stream>>>(IW, out);
  const int nblk = (NSEG + 15) / 16; // 313 waves, one per 16 segments
  k_norm_wmma<<<nblk, 32, 0, stream>>>(out + OUT_CHIST, CH_NB);
  k_norm_wmma<<<nblk, 32, 0, stream>>>(out + OUT_THIST, TH_NB);
}
